// RGCN_86440511799699
// MI455X (gfx1250) — matrix operand order repaired and checked by reading, not compiled, not run on hardware
//
#include <hip/hip_runtime.h>
#include <hip/hip_bf16.h>

// ---------------------------------------------------------------------------
// RGCN (2 layers) for MI455X / gfx1250, wave32 + WMMA bf16 + async LDS DMA.
//
//   sum_r mean_agg_r(x) @ rel_W[r]  ==  [N, R*128] @ [R*128, dout]
// with the per-node-type root term folded in as 3 masked 128-col blocks
// => one GEMM per layer, K = 8*128 + 3*128 = 1408, bf16 in / f32 accumulate.
//
// GEMM: block = 8 waves = 128x128 tile. B tile (8KB per k-step) staged into
// LDS via global_load_async_to_lds_b128 (ASYNCcnt), double-buffered with a
// dynamic XOR buffer select + unroll(disable) so there is exactly one loop
// body and the WMMA accumulators keep a single register home. B fragments
// are software-pipelined two-deep so DS loads overlap the WMMAs.
// ---------------------------------------------------------------------------

typedef __attribute__((ext_vector_type(16))) __bf16 v16bf;
typedef __attribute__((ext_vector_type(8)))  float  v8f;

union FragBF {            // 16 bf16 = 32 bytes = 8 VGPRs, loaded as 2x b128
    v16bf  v;
    float4 f[2];
};

#define RGCN_K    1408    // 8*128 rel + 3*128 root
#define RGCN_KT   44      // K / 32
#define RGCN_DIN  128
#define RGCN_R    8
#define RGCN_T    3

__device__ __forceinline__ unsigned short f2bf(float f) {
    unsigned u = __float_as_uint(f);
    u += 0x7FFFu + ((u >> 16) & 1u);       // round-to-nearest-even
    return (unsigned short)(u >> 16);
}

// ---------------------------------------------------------------- zero -----
__global__ void zero_f(float* __restrict__ p, long n) {
    long i = (long)blockIdx.x * blockDim.x + threadIdx.x;
    if (i < n) p[i] = 0.0f;
}

// ------------------------------------------------- pack weights to bf16 ----
// Packed layout: element index = ((kt*NT + nt)*32 + lane)*16 + i
// lane l, elem i  ->  K = kt*32 + 8*(l>>4) + (i&7) + 16*(i>>3) ,  n = nt*16 + (l&15)
__global__ void pack_w(const float* __restrict__ relW,
                       const float* __restrict__ rootW,
                       unsigned short* __restrict__ Wp,
                       int dout, int NT) {
    long tid   = (long)blockIdx.x * blockDim.x + threadIdx.x;
    long total = (long)RGCN_KT * NT * 512;
    if (tid >= total) return;
    int  i    = (int)(tid & 15);
    int  lane = (int)((tid >> 4) & 31);
    long rest = tid >> 9;
    int  nt   = (int)(rest % NT);
    int  kt   = (int)(rest / NT);
    int  k    = kt * 32 + ((lane >> 4) << 3) + (i & 7) + ((i >> 3) << 4);
    int  n    = nt * 16 + (lane & 15);
    float v = 0.0f;
    if (n < dout && k < RGCN_K) {
        v = (k < RGCN_R * RGCN_DIN)
                ? relW [(size_t)k * dout + n]
                : rootW[(size_t)(k - RGCN_R * RGCN_DIN) * dout + n];
    }
    Wp[tid] = f2bf(v);
}

// ------------------------------------------------- gather/select input -----
__global__ void gather_x(const float* __restrict__ x0,
                         const float* __restrict__ x1,
                         const float* __restrict__ emb2,
                         const int* __restrict__ ntype,
                         const int* __restrict__ lidx,
                         float* __restrict__ xf, int N) {
    long tid = (long)blockIdx.x * blockDim.x + threadIdx.x;
    if (tid >= (long)N * 32) return;
    int node = (int)(tid >> 5);
    int c    = (int)(tid & 31) << 2;
    int t    = ntype[node];
    const float* src = (t == 0 ? x0 : (t == 1 ? x1 : emb2));
    src += (size_t)lidx[node] * RGCN_DIN + c;
    *(float4*)(xf + (size_t)node * RGCN_DIN + c) = *(const float4*)src;
}

// ------------------------------------------------- edge scatter (atomic) ---
__global__ void __launch_bounds__(256)
scatter_edges(const float* __restrict__ X,
              const int* __restrict__ ei,   // [2,E] flat
              const int* __restrict__ et,
              float* __restrict__ Aacc,     // [N, 1024]
              float* __restrict__ cnt,      // [N, 8]
              int E, int addCnt) {
    int  lane   = threadIdx.x & 31;
    long wave   = (long)blockIdx.x * 8 + (threadIdx.x >> 5);
    long stride = (long)gridDim.x * 8;
    for (long e = wave; e < E; e += stride) {
        int src = ei[e];
        int dst = ei[(size_t)E + e];
        int r   = et[e];
        float4 vv = *(const float4*)(X + (size_t)src * RGCN_DIN + lane * 4);
        float* p  = Aacc + (size_t)dst * (RGCN_R * RGCN_DIN)
                         + (size_t)r * RGCN_DIN + lane * 4;
        atomicAdd(p + 0, vv.x);
        atomicAdd(p + 1, vv.y);
        atomicAdd(p + 2, vv.z);
        atomicAdd(p + 3, vv.w);
        if (addCnt && lane == 0) atomicAdd(cnt + (size_t)dst * RGCN_R + r, 1.0f);
    }
}

// --------------------------------- normalize + mask + convert to bf16 A ----
__global__ void build_abf(const float* __restrict__ Aacc,
                          const float* __restrict__ cnt,
                          const float* __restrict__ Xr,
                          const int* __restrict__ ntype,
                          unsigned short* __restrict__ Abf, int N) {
    long tid = (long)blockIdx.x * blockDim.x + threadIdx.x;
    if (tid >= (long)N * (RGCN_K / 4)) return;
    int row = (int)(tid / (RGCN_K / 4));
    int kq  = (int)(tid % (RGCN_K / 4)) * 4;
    float v[4];
    if (kq < RGCN_R * RGCN_DIN) {
        int   r = kq >> 7;
        float c = cnt[(size_t)row * RGCN_R + r];
        float s = (c > 0.0f) ? (1.0f / c) : 0.0f;
        const float* p = Aacc + (size_t)row * (RGCN_R * RGCN_DIN) + kq;
        v[0] = p[0] * s; v[1] = p[1] * s; v[2] = p[2] * s; v[3] = p[3] * s;
    } else {
        int   t = (kq - RGCN_R * RGCN_DIN) >> 7;
        float m = (ntype[row] == t) ? 1.0f : 0.0f;
        const float* p = Xr + (size_t)row * RGCN_DIN + (kq & 127);
        v[0] = p[0] * m; v[1] = p[1] * m; v[2] = p[2] * m; v[3] = p[3] * m;
    }
    union { unsigned short u[4]; uint2 q; } o;
    o.u[0] = f2bf(v[0]); o.u[1] = f2bf(v[1]);
    o.u[2] = f2bf(v[2]); o.u[3] = f2bf(v[3]);
    *(uint2*)(Abf + (size_t)row * RGCN_K + kq) = o.q;
}

// ------------------------------------------------------- WMMA bf16 GEMM ----
__global__ void __launch_bounds__(256)
rgcn_gemm(const unsigned short* __restrict__ A,
          const unsigned short* __restrict__ Wp,
          const float* __restrict__ bias,
          const int* __restrict__ ntype,
          float* __restrict__ out,
          int M, int NTtot, int ldout, int boundN, int ldb,
          int relu, int mblocks) {
    __shared__ unsigned short Bsh[2][4096];   // 2 x 8KB double buffer

    const int tid  = threadIdx.x;
    const int lane = tid & 31;
    const int half = lane >> 4;
    const int wv   = tid >> 5;
    const int mb   = (int)(blockIdx.x % mblocks);
    const int cg   = (int)(blockIdx.x / mblocks);
    const int m0   = mb * 128 + wv * 16;
    const int ntbase = cg * 8;

    int arow = m0 + (lane & 15);
    if (arow >= M) arow = M - 1;
    const unsigned short* aptr = A + (size_t)arow * RGCN_K + 8 * half;

    // 32-bit LDS byte address of this thread's 32B staging slot (buffer 0)
    const unsigned ldsbase = (unsigned)(size_t)(&Bsh[0][0]) + (unsigned)tid * 32u;

    // per-thread W sources: slice 0 (wrap target) and running pointer
    const unsigned short* wbase = Wp + (size_t)ntbase * 512 + (size_t)tid * 16;
    const unsigned short* wsrc  = wbase + (size_t)NTtot * 512;   // slice 1

    // prologue: stage slice 0 into buffer 0
    asm volatile("global_load_async_to_lds_b128 %0, %1, off"
                 :: "v"(ldsbase), "v"(wbase) : "memory");
    asm volatile("global_load_async_to_lds_b128 %0, %1, off offset:16"
                 :: "v"(ldsbase), "v"(wbase) : "memory");
    asm volatile("s_wait_asynccnt 0x0" ::: "memory");
    __syncthreads();

    unsigned bsel = 0;                         // byte offset of current buffer
    v8f acc[8] = {};
#pragma clang loop unroll(disable)
    for (int kt = 0; kt < RGCN_KT; ++kt) {
        // stage next slice into the other buffer (wraps to slice 0 at the end
        // -- one harmless extra 8KB DMA keeps the body uniform)
        unsigned dst = ldsbase + (bsel ^ 8192u);
        asm volatile("global_load_async_to_lds_b128 %0, %1, off"
                     :: "v"(dst), "v"(wsrc) : "memory");
        asm volatile("global_load_async_to_lds_b128 %0, %1, off offset:16"
                     :: "v"(dst), "v"(wsrc) : "memory");
        wsrc = (kt == RGCN_KT - 2) ? wbase : (wsrc + (size_t)NTtot * 512);

        FragBF a;                              // A frag: 2x b128 from global
        a.f[0] = *(const float4*)(aptr);       // K: kt*32 + 8*half   .. +8
        a.f[1] = *(const float4*)(aptr + 16);  // K: kt*32 + 16+8*half .. +8
        aptr += 32;
        __builtin_prefetch(aptr, 0, 3);        // speculative, safe

        const unsigned short* bsh = &Bsh[0][0] + (bsel >> 1) + (size_t)lane * 16;

        // two-deep software pipeline over the 8 n-tiles: load frag j+1
        // while WMMA j executes (two register homes => partial DS waits)
        FragBF b[2];
        b[0].f[0] = *(const float4*)(bsh);
        b[0].f[1] = *(const float4*)(bsh + 8);
#pragma unroll
        for (int j = 0; j < 8; ++j) {
            const int cur = j & 1;
            if (j < 7) {
                b[cur ^ 1].f[0] = *(const float4*)(bsh + (j + 1) * 512);
                b[cur ^ 1].f[1] = *(const float4*)(bsh + (j + 1) * 512 + 8);
            }
            acc[j] = __builtin_amdgcn_wmma_f32_16x16x32_bf16(
                false, a.v, false, b[cur].v, (short)0, acc[j], false, false);
        }
        bsel ^= 8192u;
        asm volatile("s_wait_asynccnt 0x0" ::: "memory");
        __syncthreads();
    }

    // Epilogue: C/D layout => lane l, vgpr v : row = m0+v+8*half, col low 4b
#pragma unroll
    for (int j = 0; j < 8; ++j) {
        int col = (ntbase + j) * 16 + (lane & 15);
        if (col >= boundN) continue;
#pragma unroll
        for (int v = 0; v < 8; ++v) {
            int row = m0 + v + 8 * half;
            if (row >= M) continue;
            float val = acc[j][v] + bias[(size_t)ntype[row] * ldb + col];
            if (relu) val = fmaxf(val, 0.0f);
            out[(size_t)row * ldout + col] = val;
        }
    }
}

// ------------------------------------------------------------ log-softmax --
__global__ void __launch_bounds__(256)
logsoftmax_rows(const float* __restrict__ logits, float* __restrict__ out,
                int C, int ld) {
    __shared__ float red[256];
    int row = blockIdx.x;
    const float* in = logits + (size_t)row * ld;
    float m = -INFINITY;
    for (int j = threadIdx.x; j < C; j += 256) m = fmaxf(m, in[j]);
    red[threadIdx.x] = m; __syncthreads();
    for (int s = 128; s > 0; s >>= 1) {
        if (threadIdx.x < s) red[threadIdx.x] = fmaxf(red[threadIdx.x], red[threadIdx.x + s]);
        __syncthreads();
    }
    m = red[0]; __syncthreads();
    float sum = 0.0f;
    for (int j = threadIdx.x; j < C; j += 256) sum += __expf(in[j] - m);
    red[threadIdx.x] = sum; __syncthreads();
    for (int s = 128; s > 0; s >>= 1) {
        if (threadIdx.x < s) red[threadIdx.x] += red[threadIdx.x + s];
        __syncthreads();
    }
    float lse = m + __logf(red[0]);
    for (int j = threadIdx.x; j < C; j += 256)
        out[(size_t)row * C + j] = in[j] - lse;
}

// ---------------------------------------------------------------------------
extern "C" void kernel_launch(void* const* d_in, const int* in_sizes, int n_in,
                              void* d_out, int out_size, void* d_ws, size_t ws_size,
                              hipStream_t stream) {
    const float* x0     = (const float*)d_in[0];
    const float* x1     = (const float*)d_in[1];
    const float* emb2   = (const float*)d_in[2];
    const float* relW1  = (const float*)d_in[3];
    const float* rootW1 = (const float*)d_in[4];
    const float* rootb1 = (const float*)d_in[5];
    const float* relW2  = (const float*)d_in[6];
    const float* rootW2 = (const float*)d_in[7];
    const float* rootb2 = (const float*)d_in[8];
    const int*   eidx   = (const int*)d_in[9];
    const int*   etype  = (const int*)d_in[10];
    const int*   ntype  = (const int*)d_in[11];
    const int*   lidx   = (const int*)d_in[12];

    const int E = in_sizes[9] / 2;
    const int N = in_sizes[11];
    const int dout1 = in_sizes[3] / (RGCN_R * RGCN_DIN);   // 128
    const int dout2 = in_sizes[6] / (RGCN_R * RGCN_DIN);   // 349
    const int NP1 = ((dout1 + 127) / 128) * 128;           // 128
    const int NP2 = ((dout2 + 127) / 128) * 128;           // 384
    const int NT1 = NP1 / 16, NT2 = NP2 / 16;
    const int CG1 = NP1 / 128, CG2 = NP2 / 128;

    // workspace layout
    char* w = (char*)d_ws;
    float* Aacc = (float*)w;                  w += (size_t)N * 1024 * 4;  // + logits [N,384]
    unsigned short* Abf = (unsigned short*)w; w += (size_t)N * RGCN_K * 2;
    float* xf  = (float*)w;                   w += (size_t)N * RGCN_DIN * 4; // x, then h
    float* cnt = (float*)w;                   w += (size_t)N * RGCN_R * 4;
    unsigned short* W1p = (unsigned short*)w; w += (size_t)RGCN_KT * NT1 * 512 * 2;
    unsigned short* W2p = (unsigned short*)w;

    const int TB = 256;
    long nz1 = (long)N * 1024;
    long nzc = (long)N * RGCN_R;

    // 0) zero accumulators + counts
    zero_f<<<(int)((nz1 + TB - 1) / TB), TB, 0, stream>>>(Aacc, nz1);
    zero_f<<<(int)((nzc + TB - 1) / TB), TB, 0, stream>>>(cnt, nzc);

    // 1) pack weights (bf16 fragment order), both layers
    long tw1 = (long)RGCN_KT * NT1 * 512;
    long tw2 = (long)RGCN_KT * NT2 * 512;
    pack_w<<<(int)((tw1 + TB - 1) / TB), TB, 0, stream>>>(relW1, rootW1, W1p, dout1, NT1);
    pack_w<<<(int)((tw2 + TB - 1) / TB), TB, 0, stream>>>(relW2, rootW2, W2p, dout2, NT2);

    // 2) gather/select node features
    long tg = (long)N * 32;
    gather_x<<<(int)((tg + TB - 1) / TB), TB, 0, stream>>>(x0, x1, emb2, ntype, lidx, xf, N);

    // 3) layer-1 edge scatter (+degree counts)
    int sgrid = (E + 7) / 8;
    scatter_edges<<<sgrid, TB, 0, stream>>>(xf, eidx, etype, Aacc, cnt, E, 1);

    // 4) build bf16 A1 (mean-normalized + masked root blocks)
    long ta = (long)N * (RGCN_K / 4);
    build_abf<<<(int)((ta + TB - 1) / TB), TB, 0, stream>>>(Aacc, cnt, xf, ntype, Abf, N);

    // 5) GEMM1 + bias + ReLU -> h (reuses xf buffer)
    int mblocks = (N + 127) / 128;
    rgcn_gemm<<<mblocks * CG1, TB, 0, stream>>>(Abf, W1p, rootb1, ntype, xf,
                                                N, NT1, dout1, dout1, dout1, 1, mblocks);

    // 6) zero accumulators, layer-2 scatter of h
    zero_f<<<(int)((nz1 + TB - 1) / TB), TB, 0, stream>>>(Aacc, nz1);
    scatter_edges<<<sgrid, TB, 0, stream>>>(xf, eidx, etype, Aacc, cnt, E, 0);

    // 7) build bf16 A2
    build_abf<<<(int)((ta + TB - 1) / TB), TB, 0, stream>>>(Aacc, cnt, xf, ntype, Abf, N);

    // 8) GEMM2 + bias -> logits (reuses Aacc as [N, NP2] f32)
    float* logits = Aacc;
    rgcn_gemm<<<mblocks * CG2, TB, 0, stream>>>(Abf, W2p, rootb2, ntype, logits,
                                                N, NT2, NP2, dout2, dout2, 0, mblocks);

    // 9) row-wise log-softmax -> d_out [N, dout2]
    logsoftmax_rows<<<N, TB, 0, stream>>>(logits, (float*)d_out, dout2, NP2);
}
